// MathSDPDecodeAttention_67242007986746
// MI455X (gfx1250) — matrix operand order repaired and checked by reading, not compiled, not run on hardware
//
#include <hip/hip_runtime.h>
#include <stdint.h>

// Causal flash-attention forward, B=2,H=16,S=2048,D=128, fp32 in/out.
// Pre-pass: Q (scaled by 1/sqrt(D)*log2e), K -> bf16; V -> bf16 transposed.
// Main kernel (transposed formulation): S^T = K*Q^T, O^T = V^T*P^T, bf16 WMMA
// with f32 accumulate, per-lane softmax, async double-buffered LDS staging.

static constexpr int kS  = 2048;
static constexpr int kD  = 128;
static constexpr int kBH = 32;       // B*H
static constexpr int kBM = 64;       // query rows per workgroup (4 waves x 16)
static constexpr int kBN = 64;       // keys per iteration

typedef __attribute__((ext_vector_type(16))) __bf16 v16bf;
typedef __attribute__((ext_vector_type(8)))  __bf16 v8bf;
typedef __attribute__((ext_vector_type(8)))  float  v8f;

__device__ __forceinline__ unsigned short f2bf(float f) {      // RNE
  unsigned u = __builtin_bit_cast(unsigned, f);
  unsigned r = u + 0x7FFFu + ((u >> 16) & 1u);
  return (unsigned short)(r >> 16);
}
__device__ __forceinline__ unsigned pk2(float lo, float hi) {
  return (unsigned)f2bf(lo) | ((unsigned)f2bf(hi) << 16);
}
// pack two floats as truncated bf16 pair with a single v_perm_b32
__device__ __forceinline__ unsigned pack_bf16_trunc(float lo, float hi) {
  return __builtin_amdgcn_perm(__builtin_bit_cast(unsigned, hi),
                               __builtin_bit_cast(unsigned, lo), 0x07060302u);
}

// ---------------- pre-pass 1: Q (scaled) and K -> bf16 -----------------------
__global__ __launch_bounds__(256)
void cvt_qk_kernel(const float* __restrict__ q, const float* __restrict__ k,
                   unsigned short* __restrict__ qbf, unsigned short* __restrict__ kbf) {
  const float qs = 0.08838834764831845f * 1.4426950408889634f; // scale * log2(e)
  size_t base = ((size_t)blockIdx.x * 256 + threadIdx.x) * 8;
  float4 a0 = *(const float4*)(q + base);
  float4 a1 = *(const float4*)(q + base + 4);
  uint4 qo;
  qo.x = pk2(a0.x * qs, a0.y * qs); qo.y = pk2(a0.z * qs, a0.w * qs);
  qo.z = pk2(a1.x * qs, a1.y * qs); qo.w = pk2(a1.z * qs, a1.w * qs);
  *(uint4*)(qbf + base) = qo;
  float4 b0 = *(const float4*)(k + base);
  float4 b1 = *(const float4*)(k + base + 4);
  uint4 ko;
  ko.x = pk2(b0.x, b0.y); ko.y = pk2(b0.z, b0.w);
  ko.z = pk2(b1.x, b1.y); ko.w = pk2(b1.z, b1.w);
  *(uint4*)(kbf + base) = ko;
}

// ---------------- pre-pass 2: V -> bf16 transposed [bh][d][key] --------------
__global__ __launch_bounds__(128)
void vtrans_kernel(const float* __restrict__ v, unsigned short* __restrict__ vt) {
  __shared__ unsigned short T[64 * kD];               // 16 KB
  const int tid = threadIdx.x;
  const int kb  = blockIdx.x * 64;
  const int bh  = blockIdx.y;
  const float* vp = v + ((size_t)bh * kS + kb) * kD;
  for (int i = tid; i < 64 * kD / 4; i += 128) {
    float4 f = ((const float4*)vp)[i];
    T[i * 4 + 0] = f2bf(f.x); T[i * 4 + 1] = f2bf(f.y);
    T[i * 4 + 2] = f2bf(f.z); T[i * 4 + 3] = f2bf(f.w);
  }
  __syncthreads();
  const int d = tid;
  unsigned short* op = vt + ((size_t)bh * kD + d) * kS + kb;
#pragma unroll
  for (int g = 0; g < 8; ++g) {
    unsigned w[4];
#pragma unroll
    for (int p = 0; p < 4; ++p) {
      unsigned short e0 = T[(g * 8 + p * 2 + 0) * kD + d];
      unsigned short e1 = T[(g * 8 + p * 2 + 1) * kD + d];
      w[p] = (unsigned)e0 | ((unsigned)e1 << 16);
    }
    *(uint4*)(op + g * 8) = make_uint4(w[0], w[1], w[2], w[3]);
  }
}

// ---------------- async 16B/lane copy, tracked by ASYNCcnt -------------------
__device__ __forceinline__ void async_cp16(unsigned lds, const char* g, int imm) {
  switch (imm) {
    case 0:   asm volatile("global_load_async_to_lds_b128 %0, %1, off"            :: "v"(lds), "v"(g) : "memory"); break;
    case 16:  asm volatile("global_load_async_to_lds_b128 %0, %1, off offset:16"  :: "v"(lds), "v"(g) : "memory"); break;
    case 32:  asm volatile("global_load_async_to_lds_b128 %0, %1, off offset:32"  :: "v"(lds), "v"(g) : "memory"); break;
    case 48:  asm volatile("global_load_async_to_lds_b128 %0, %1, off offset:48"  :: "v"(lds), "v"(g) : "memory"); break;
    case 64:  asm volatile("global_load_async_to_lds_b128 %0, %1, off offset:64"  :: "v"(lds), "v"(g) : "memory"); break;
    case 80:  asm volatile("global_load_async_to_lds_b128 %0, %1, off offset:80"  :: "v"(lds), "v"(g) : "memory"); break;
    case 96:  asm volatile("global_load_async_to_lds_b128 %0, %1, off offset:96"  :: "v"(lds), "v"(g) : "memory"); break;
    default:  asm volatile("global_load_async_to_lds_b128 %0, %1, off offset:112" :: "v"(lds), "v"(g) : "memory"); break;
  }
}

// ---------------- main attention kernel (transposed, bf16 inputs) ------------
__global__ __launch_bounds__(128)
void fa_bf16_kernel(const unsigned short* __restrict__ qbf,
                    const unsigned short* __restrict__ kbf,
                    const unsigned short* __restrict__ vtbf,
                    float* __restrict__ out) {
  __shared__ unsigned short Ks[2][kBN * kD];   // 2 x 16 KB, [key][d]
  __shared__ unsigned short VTs[2][kD * kBN];  // 2 x 16 KB, [d][key]
  __shared__ unsigned short PsT[kBM * kBN];    // 8 KB, P^T [query][key]

  const int tid   = threadIdx.x;
  const int lane  = tid & 31;
  const int wave  = tid >> 5;
  const int row16 = lane & 15;   // this lane's query column
  const int hi    = lane >> 4;
  const int qbase = blockIdx.x * kBM;
  const int bh    = blockIdx.y;

  // ---- Q as B-fragments (lane = query column, 32B contiguous per chunk) ----
  const unsigned short* qg =
      qbf + ((size_t)(bh * kS + qbase + wave * 16 + row16)) * kD;
  v16bf qb[4];
#pragma unroll
  for (int c = 0; c < 4; ++c)
    qb[c] = *(const v16bf*)(qg + c * 32 + hi * 16);

  v8f acc[8];                       // O^T accumulator: 8 d-chunks (d x query)
  v8f zero = {};
#pragma unroll
  for (int dc = 0; dc < 8; ++dc) acc[dc] = zero;
  float mrow = -1e30f, lrow = 0.0f; // per-lane (per-query) softmax state
  const int qcol = qbase + wave * 16 + row16;
  const int nj = qbase / kBN + 1;

  // Per-thread copy bases: global and LDS advance +16B per chunk together.
  const char* kptr = (const char*)(kbf + (size_t)bh * kS * kD) + tid * 128;
  const char* vptr = (const char*)(vtbf + (size_t)bh * kD * kS) + tid * 4096;
  const unsigned ldsK = (unsigned)(uintptr_t)&Ks[0][0] + (unsigned)tid * 128u;
  const unsigned ldsV = (unsigned)(uintptr_t)&VTs[0][0] + (unsigned)tid * 128u;

  auto stage = [&](int buf, const char* kp, const char* vp) {
#pragma unroll
    for (int c = 0; c < 8; ++c)
      async_cp16(ldsK + (unsigned)buf * (kBN * kD * 2), kp, c * 16);
#pragma unroll
    for (int c = 0; c < 8; ++c)
      async_cp16(ldsV + (unsigned)buf * (kBN * kD * 2), vp, c * 16);
  };

  stage(0, kptr, vptr);

  for (int j = 0; j < nj; ++j) {
    asm volatile("s_wait_asynccnt 0" ::: "memory");
    __syncthreads();
    const int buf = j & 1;
    if (j + 1 < nj) stage(buf ^ 1, kptr + kBN * kD * 2, vptr + kBN * 2);
    kptr += kBN * kD * 2;
    vptr += kBN * 2;
    const unsigned short* KsB  = &Ks[buf][0];
    const unsigned short* VTsB = &VTs[buf][0];
    const int kbase = j * kBN;
    const bool domask = (j == nj - 1);

    // ---- S^T = K * Q^T : four 16(key)x16(query) tiles ----
    v8f st[4];
#pragma unroll
    for (int t = 0; t < 4; ++t) st[t] = zero;
#pragma unroll
    for (int c = 0; c < 4; ++c) {
      v16bf ka[4];
#pragma unroll
      for (int t = 0; t < 4; ++t) {
        const unsigned short* ab = &KsB[(t * 16 + row16) * kD + c * 32 + hi * 16];
        v8bf lo = *(const v8bf*)ab;
        v8bf h8 = *(const v8bf*)(ab + 8);
        ka[t] = __builtin_shufflevector(lo, h8, 0,1,2,3,4,5,6,7,8,9,10,11,12,13,14,15);
      }
#pragma unroll
      for (int t = 0; t < 4; ++t)
        st[t] = __builtin_amdgcn_wmma_f32_16x16x32_bf16(
                    false, ka[t], false, qb[c], (short)0, st[t], false, false);
    }

    // ---- causal mask (last block only): key = kbase+t*16+hi*8+r ----
    if (domask) {
#pragma unroll
      for (int t = 0; t < 4; ++t)
#pragma unroll
        for (int r = 0; r < 8; ++r)
          if (kbase + t * 16 + hi * 8 + r > qcol) st[t][r] = -1e30f;
    }

    // ---- per-lane online softmax over this lane's 32 scores ----
    v8f vm = st[0];
#pragma unroll
    for (int t = 1; t < 4; ++t)
#pragma unroll
      for (int r = 0; r < 8; ++r) vm[r] = fmaxf(vm[r], st[t][r]);
    float mx = fmaxf(fmaxf(fmaxf(vm[0], vm[1]), fmaxf(vm[2], vm[3])),
                     fmaxf(fmaxf(vm[4], vm[5]), fmaxf(vm[6], vm[7])));
    mx = fmaxf(mx, __shfl_xor(mx, 16));       // combine lane pair (n, n+16)
    const float mnew = fmaxf(mrow, mx);
#pragma unroll
    for (int t = 0; t < 4; ++t)
#pragma unroll
      for (int r = 0; r < 8; ++r)
        st[t][r] = __builtin_amdgcn_exp2f(st[t][r] - mnew);
    v8f vs = st[0];
#pragma unroll
    for (int t = 1; t < 4; ++t)
#pragma unroll
      for (int r = 0; r < 8; ++r) vs[r] += st[t][r];
    float sum = (((vs[0] + vs[1]) + (vs[2] + vs[3])) +
                 ((vs[4] + vs[5]) + (vs[6] + vs[7])));
    sum += __shfl_xor(sum, 16);
    const float corr = __builtin_amdgcn_exp2f(mrow - mnew);
    lrow = lrow * corr + sum;
    mrow = mnew;
#pragma unroll
    for (int dc = 0; dc < 8; ++dc) acc[dc] *= corr;

    // ---- pack P^T (truncated bf16) and spill: 4 x ds_store_b128 ----
#pragma unroll
    for (int t = 0; t < 4; ++t) {
      uint4 w;
      w.x = pack_bf16_trunc(st[t][0], st[t][1]);
      w.y = pack_bf16_trunc(st[t][2], st[t][3]);
      w.z = pack_bf16_trunc(st[t][4], st[t][5]);
      w.w = pack_bf16_trunc(st[t][6], st[t][7]);
      *(uint4*)&PsT[(wave * 16 + row16) * kBN + t * 16 + hi * 8] = w;
    }

    asm volatile("" ::: "memory");  // per-wave LDS ops are in-order

    // ---- P^T as B-fragments (keys 0..31 / 32..63), 32B contiguous ----
    v16bf pt[2];
#pragma unroll
    for (int kc = 0; kc < 2; ++kc)
      pt[kc] = *(const v16bf*)&PsT[(wave * 16 + row16) * kBN + kc * 32 + hi * 16];

    // ---- O^T += V^T * P^T ----
#pragma unroll
    for (int kc = 0; kc < 2; ++kc) {
#pragma unroll
      for (int dc = 0; dc < 8; ++dc) {
        const unsigned short* ab = &VTsB[(dc * 16 + row16) * kBN + kc * 32 + hi * 16];
        v8bf lo = *(const v8bf*)ab;
        v8bf h8 = *(const v8bf*)(ab + 8);
        v16bf va = __builtin_shufflevector(lo, h8, 0,1,2,3,4,5,6,7,8,9,10,11,12,13,14,15);
        acc[dc] = __builtin_amdgcn_wmma_f32_16x16x32_bf16(
                      false, va, false, pt[kc], (short)0, acc[dc], false, false);
      }
    }
  }

  // ---- epilogue: each lane owns one output row; 2 x b128 stores per chunk ----
  const float inv = 1.0f / lrow;
  float* op = out + (size_t)bh * kS * kD + (size_t)qcol * kD + hi * 8;
#pragma unroll
  for (int dc = 0; dc < 8; ++dc) {
    float4 lo4 = make_float4(acc[dc][0] * inv, acc[dc][1] * inv,
                             acc[dc][2] * inv, acc[dc][3] * inv);
    float4 hi4 = make_float4(acc[dc][4] * inv, acc[dc][5] * inv,
                             acc[dc][6] * inv, acc[dc][7] * inv);
    *(float4*)(op + dc * 16)     = lo4;
    *(float4*)(op + dc * 16 + 4) = hi4;
  }
}

// ---------------- fallback (fp32 global, no workspace) -----------------------
__global__ __launch_bounds__(128)
void fa_fallback_kernel(const float* __restrict__ q, const float* __restrict__ k,
                        const float* __restrict__ v, float* __restrict__ out) {
  __shared__ unsigned short Qs[kBM * kD];
  __shared__ unsigned short Ksf[32 * kD];
  __shared__ unsigned short VTsf[kD * 32];
  __shared__ unsigned short Psf[kBM * 32];
  const int tid = threadIdx.x, lane = tid & 31, wave = tid >> 5;
  const int row16 = lane & 15, hi = lane >> 4;
  const int qbase = blockIdx.x * kBM, bh = blockIdx.y;
  const float scale = 0.08838834764831845f;
  const float* qp = q + (size_t)bh * kS * kD;
  const float* kp = k + (size_t)bh * kS * kD;
  const float* vp = v + (size_t)bh * kS * kD;
  float* op = out + (size_t)bh * kS * kD;
  for (int i = tid; i < kBM * kD / 4; i += 128) {
    float4 f = ((const float4*)(qp + (size_t)qbase * kD))[i];
    Qs[i*4+0] = f2bf(f.x*scale); Qs[i*4+1] = f2bf(f.y*scale);
    Qs[i*4+2] = f2bf(f.z*scale); Qs[i*4+3] = f2bf(f.w*scale);
  }
  __syncthreads();
  v16bf qa[4];
#pragma unroll
  for (int c = 0; c < 4; ++c) {
    const unsigned short* base = &Qs[(wave*16 + row16)*kD + c*32 + hi*16];
    v8bf lo = *(const v8bf*)base; v8bf h8 = *(const v8bf*)(base + 8);
    qa[c] = __builtin_shufflevector(lo, h8, 0,1,2,3,4,5,6,7,8,9,10,11,12,13,14,15);
  }
  v8f acc[8]; v8f zero = {};
#pragma unroll
  for (int dc = 0; dc < 8; ++dc) acc[dc] = zero;
  float mrow[8], lrow[8];
#pragma unroll
  for (int r = 0; r < 8; ++r) { mrow[r] = -1e30f; lrow[r] = 0.0f; }
  const int mglob0 = qbase + wave*16 + hi*8;
  const int nj = qbase / 32 + 2;
  for (int j = 0; j < nj; ++j) {
    const int kbase = j * 32;
    __syncthreads();
    for (int i = tid; i < 32 * kD / 4; i += 128) {
      float4 f = ((const float4*)(kp + (size_t)kbase*kD))[i];
      Ksf[i*4+0]=f2bf(f.x); Ksf[i*4+1]=f2bf(f.y); Ksf[i*4+2]=f2bf(f.z); Ksf[i*4+3]=f2bf(f.w);
      float4 g = ((const float4*)(vp + (size_t)kbase*kD))[i];
      int key = (i*4)/kD, d0 = (i*4)%kD;
      VTsf[(d0+0)*32+key]=f2bf(g.x); VTsf[(d0+1)*32+key]=f2bf(g.y);
      VTsf[(d0+2)*32+key]=f2bf(g.z); VTsf[(d0+3)*32+key]=f2bf(g.w);
    }
    __syncthreads();
    v8f st[2];
#pragma unroll
    for (int t = 0; t < 2; ++t) {
      v8f s = zero;
#pragma unroll
      for (int c = 0; c < 4; ++c) {
        v16bf bf = *(const v16bf*)&Ksf[(t*16+row16)*kD + c*32 + hi*16];
        s = __builtin_amdgcn_wmma_f32_16x16x32_bf16(false, qa[c], false, bf, (short)0, s, false, false);
      }
      st[t] = s;
    }
#pragma unroll
    for (int r = 0; r < 8; ++r) {
      const int mg = mglob0 + r;
      float s0 = st[0][r]; if (kbase + row16 > mg) s0 = -1e30f;
      float s1 = st[1][r]; if (kbase + 16 + row16 > mg) s1 = -1e30f;
      float mx = fmaxf(s0, s1);
      mx = fmaxf(mx, __shfl_xor(mx,1)); mx = fmaxf(mx, __shfl_xor(mx,2));
      mx = fmaxf(mx, __shfl_xor(mx,4)); mx = fmaxf(mx, __shfl_xor(mx,8));
      const float mnew = fmaxf(mrow[r], mx);
      const float p0 = __expf(s0-mnew), p1 = __expf(s1-mnew), corr = __expf(mrow[r]-mnew);
      float rs = p0 + p1;
      rs += __shfl_xor(rs,1); rs += __shfl_xor(rs,2);
      rs += __shfl_xor(rs,4); rs += __shfl_xor(rs,8);
      lrow[r] = lrow[r]*corr + rs; mrow[r] = mnew;
#pragma unroll
      for (int dc = 0; dc < 8; ++dc) acc[dc][r] *= corr;
      const int prow = wave*16 + r + hi*8;
      Psf[prow*32 + row16] = f2bf(p0); Psf[prow*32 + 16 + row16] = f2bf(p1);
    }
    asm volatile("" ::: "memory");
    const unsigned short* pb = &Psf[(wave*16+row16)*32 + hi*16];
    v8bf plo = *(const v8bf*)pb; v8bf phi = *(const v8bf*)(pb+8);
    v16bf pf = __builtin_shufflevector(plo, phi, 0,1,2,3,4,5,6,7,8,9,10,11,12,13,14,15);
#pragma unroll
    for (int dc = 0; dc < 8; ++dc) {
      v16bf vf = *(const v16bf*)&VTsf[(dc*16+row16)*32 + hi*16];
      acc[dc] = __builtin_amdgcn_wmma_f32_16x16x32_bf16(false, pf, false, vf, (short)0, acc[dc], false, false);
    }
  }
#pragma unroll
  for (int r = 0; r < 8; ++r) {
    const float inv = 1.0f / lrow[r];
    const int row = qbase + wave*16 + r + hi*8;
#pragma unroll
    for (int dc = 0; dc < 8; ++dc)
      op[(size_t)row*kD + dc*16 + row16] = acc[dc][r] * inv;
  }
}

extern "C" void kernel_launch(void* const* d_in, const int* in_sizes, int n_in,
                              void* d_out, int out_size, void* d_ws, size_t ws_size,
                              hipStream_t stream) {
  (void)in_sizes; (void)n_in; (void)out_size;
  const float* q = (const float*)d_in[0];
  const float* k = (const float*)d_in[1];
  const float* v = (const float*)d_in[2];
  float* out = (float*)d_out;
  const size_t nElem = (size_t)kBH * kS * kD;          // 8,388,608
  const size_t need  = 3 * nElem * sizeof(unsigned short);
  if (ws_size >= need) {
    unsigned short* qbf = (unsigned short*)d_ws;
    unsigned short* kbf = qbf + nElem;
    unsigned short* vt  = kbf + nElem;
    cvt_qk_kernel<<<dim3((unsigned)(nElem / (256 * 8))), 256, 0, stream>>>(q, k, qbf, kbf);
    vtrans_kernel<<<dim3(kS / 64, kBH), 128, 0, stream>>>(v, vt);
    fa_bf16_kernel<<<dim3(kS / kBM, kBH), 128, 0, stream>>>(qbf, kbf, vt, out);
  } else {
    fa_fallback_kernel<<<dim3(kS / kBM, kBH), 128, 0, stream>>>(q, k, v, out);
  }
}
